// DiffuserAttention_75617194213544
// MI455X (gfx1250) — matrix operand order repaired and checked by reading, not compiled, not run on hardware
//
#include <hip/hip_runtime.h>
#include <hip/hip_bf16.h>

// ---------------- problem constants ----------------
constexpr int Bb = 2, Ss = 4096, Dd = 768, Hh = 12, HDd = 64;
constexpr int NN = Bb * Ss;            // 8192 nodes
constexpr int EE = 524288;             // edges
constexpr float ALPHA = 0.1f;
constexpr int N_STEPS = 5;
constexpr float LN_EPS = 1e-12f;
constexpr float QSCALE = 0.125f;       // 1/sqrt(64)

typedef __attribute__((ext_vector_type(16))) __bf16 v16bf;
typedef __attribute__((ext_vector_type(8)))  __bf16 v8bf;
typedef __attribute__((ext_vector_type(8)))  float  v8f;

// ---------------- helpers ----------------
__device__ inline __bf16 f2bf(float f) {
  union { unsigned u; float f; } in; in.f = f;
  unsigned u = in.u;
  u = u + 0x7FFFu + ((u >> 16) & 1u);          // RNE
  union { unsigned short s; __bf16 b; } out;
  out.s = (unsigned short)(u >> 16);
  return out.b;
}
// monotonic float<->uint encoding for atomicMax on floats
__device__ inline unsigned encF(float f) {
  unsigned u = __float_as_uint(f);
  return (u & 0x80000000u) ? ~u : (u | 0x80000000u);
}
__device__ inline float decF(unsigned e) {
  unsigned u = (e & 0x80000000u) ? (e ^ 0x80000000u) : ~e;
  return __uint_as_float(u);
}

// ---------------- elementwise / fill ----------------
__global__ __launch_bounds__(256) void k_f32_to_bf16(const float* in, __bf16* out, int n) {
  int t = blockIdx.x * 256 + threadIdx.x;
  if (t < n) out[t] = f2bf(in[t]);
}
__global__ __launch_bounds__(256) void k_transpose_bf16(const float* in, __bf16* out, int n) {
  int t = blockIdx.x * 256 + threadIdx.x;
  if (t >= n * n) return;
  int r = t / n, c = t - r * n;                // out[r][c] = in[c][r]
  out[t] = f2bf(in[c * n + r]);
}
__global__ __launch_bounds__(256) void k_fill_f32(float* p, float v, int n) {
  int t = blockIdx.x * 256 + threadIdx.x;
  if (t < n) p[t] = v;
}
__global__ __launch_bounds__(256) void k_fill_u32(unsigned* p, unsigned v, int n) {
  int t = blockIdx.x * 256 + threadIdx.x;
  if (t < n) p[t] = v;
}

// ---------------- WMMA bf16 GEMM:  C[M,Nout] = A[M,K] @ Bt[Nout,K]^T ----------------
// epilogue: C = (acc + bias[col]) * scale + (resid ? resid : 0)
__device__ inline v16bf load_a_frag(const __bf16* rowp, int k0, int half) {
  // A 16x32 bf16 frag: per lane two 8-elem runs at K = half*8 and 16+half*8
  v8bf lo = *(const v8bf*)(rowp + k0 + half * 8);
  v8bf hi = *(const v8bf*)(rowp + k0 + 16 + half * 8);
  return __builtin_shufflevector(lo, hi, 0,1,2,3,4,5,6,7,8,9,10,11,12,13,14,15);
}
__global__ __launch_bounds__(256) void k_gemm_bf16(
    const __bf16* __restrict__ A, const __bf16* __restrict__ Bt,
    const float* __restrict__ bias, const float* __restrict__ resid,
    float* __restrict__ C, int M, int Nout, int K, float scale) {
  const int lane = threadIdx.x & 31;
  const int wave = threadIdx.x >> 5;
  const int wr = wave >> 2, wc = wave & 3;           // 2x4 wave grid
  const int rowBase = blockIdx.y * 64 + wr * 32;
  const int colBase = blockIdx.x * 128 + wc * 32;
  const int r = lane & 15, half = lane >> 4;

  const __bf16* a0p = A + (size_t)(rowBase + r) * K;
  const __bf16* a1p = A + (size_t)(rowBase + 16 + r) * K;
  const __bf16* b0p = Bt + (size_t)(colBase + r) * K;
  const __bf16* b1p = Bt + (size_t)(colBase + 16 + r) * K;

  v8f acc00 = {}, acc01 = {}, acc10 = {}, acc11 = {};
  for (int k0 = 0; k0 < K; k0 += 32) {
    v16bf a0 = load_a_frag(a0p, k0, half);
    v16bf a1 = load_a_frag(a1p, k0, half);
    // B 32x16 frag: per lane two 8-elem runs at K = half*8 and 16+half*8
    v16bf b0 = load_a_frag(b0p, k0, half);
    v16bf b1 = load_a_frag(b1p, k0, half);
    acc00 = __builtin_amdgcn_wmma_f32_16x16x32_bf16(false, a0, false, b0, (short)0, acc00, false, false);
    acc01 = __builtin_amdgcn_wmma_f32_16x16x32_bf16(false, a0, false, b1, (short)0, acc01, false, false);
    acc10 = __builtin_amdgcn_wmma_f32_16x16x32_bf16(false, a1, false, b0, (short)0, acc10, false, false);
    acc11 = __builtin_amdgcn_wmma_f32_16x16x32_bf16(false, a1, false, b1, (short)0, acc11, false, false);
  }
  // C/D layout: col = lane&15, row = i + (lane>>4)*8
  const int col0 = colBase + r, col1 = col0 + 16;
  const int rb0 = rowBase + half * 8, rb1 = rb0 + 16;
  const float bv0 = bias[col0], bv1 = bias[col1];
  const bool hasR = (resid != nullptr);
#pragma unroll
  for (int i = 0; i < 8; ++i) {
    size_t o00 = (size_t)(rb0 + i) * Nout + col0;
    size_t o01 = (size_t)(rb0 + i) * Nout + col1;
    size_t o10 = (size_t)(rb1 + i) * Nout + col0;
    size_t o11 = (size_t)(rb1 + i) * Nout + col1;
    float v00 = (acc00[i] + bv0) * scale;
    float v01 = (acc01[i] + bv1) * scale;
    float v10 = (acc10[i] + bv0) * scale;
    float v11 = (acc11[i] + bv1) * scale;
    if (hasR) { v00 += resid[o00]; v01 += resid[o01]; v10 += resid[o10]; v11 += resid[o11]; }
    C[o00] = v00; C[o01] = v01; C[o10] = v10; C[o11] = v11;
  }
}

// ---------------- edge softmax ----------------
__global__ __launch_bounds__(256) void k_edge_scores(
    const float* __restrict__ q, const float* __restrict__ k,
    const int* __restrict__ src, const int* __restrict__ dst,
    const int* __restrict__ am, float* __restrict__ scores) {
  int t = blockIdx.x * 256 + threadIdx.x;
  if (t >= EE * Hh) return;
  int e = t / Hh, h = t - e * Hh;
  int sn = src[e], dn = dst[e];
  const float4* kp = (const float4*)(k + (size_t)sn * Dd + h * HDd);
  const float4* qp = (const float4*)(q + (size_t)dn * Dd + h * HDd);
  float s = 0.f;
#pragma unroll
  for (int i = 0; i < 16; ++i) {
    float4 a = kp[i], b = qp[i];
    s = fmaf(a.x, b.x, s); s = fmaf(a.y, b.y, s);
    s = fmaf(a.z, b.z, s); s = fmaf(a.w, b.w, s);
  }
  bool m = (am[sn] >= 0) && (am[dn] >= 0);
  scores[t] = m ? s : -1e4f;
}
__global__ __launch_bounds__(256) void k_edge_max(
    const float* __restrict__ scores, const int* __restrict__ dst, unsigned* __restrict__ smax) {
  int t = blockIdx.x * 256 + threadIdx.x;
  if (t >= EE * Hh) return;
  int e = t / Hh, h = t - e * Hh;
  atomicMax(&smax[(size_t)dst[e] * Hh + h], encF(scores[t]));
}
__global__ __launch_bounds__(256) void k_edge_exp(
    float* __restrict__ scores, const int* __restrict__ dst,
    const unsigned* __restrict__ smax, float* __restrict__ denom) {
  int t = blockIdx.x * 256 + threadIdx.x;
  if (t >= EE * Hh) return;
  int e = t / Hh, h = t - e * Hh;
  int dn = dst[e];
  float m = decF(smax[(size_t)dn * Hh + h]);
  if (!(m > -1e37f)) m = 0.f;                // empty-segment -> 0 (matches ref)
  float ex = __expf(scores[t] - m);
  scores[t] = ex;
  atomicAdd(&denom[(size_t)dn * Hh + h], ex);
}
__global__ __launch_bounds__(256) void k_edge_norm(
    float* __restrict__ scores, const int* __restrict__ dst, const float* __restrict__ denom) {
  int t = blockIdx.x * 256 + threadIdx.x;
  if (t >= EE * Hh) return;
  int e = t / Hh, h = t - e * Hh;
  scores[t] = scores[t] / fmaxf(denom[(size_t)dst[e] * Hh + h], 1e-9f);
}

// ---------------- sparse diffusion ----------------
// thread handles one float4 chunk of one edge: 192 chunks per row of 768
__global__ __launch_bounds__(256) void k_scatter(
    const float* __restrict__ attn, const float* __restrict__ hsrc,
    const int* __restrict__ src, const int* __restrict__ dst, float* __restrict__ hdst) {
  int t = blockIdx.x * 256 + threadIdx.x;
  if (t >= EE * 192) return;
  int e = t / 192, c = t - e * 192;
  int h = c >> 4;                               // 16 float4 chunks per head
  float a = attn[(size_t)e * Hh + h];
  int sn = src[e], dn = dst[e];
  float4 hv = ((const float4*)hsrc)[(size_t)sn * 192 + c];
  float* dp = hdst + (size_t)dn * Dd + c * 4;
  atomicAdd(dp + 0, a * hv.x);
  atomicAdd(dp + 1, a * hv.y);
  atomicAdd(dp + 2, a * hv.z);
  atomicAdd(dp + 3, a * hv.w);
}
__global__ __launch_bounds__(256) void k_combine(
    float* __restrict__ hn, const float* __restrict__ v, __bf16* __restrict__ bfout, int n) {
  int t = blockIdx.x * 256 + threadIdx.x;
  if (t >= n) return;
  float x = (1.0f - ALPHA) * hn[t] + ALPHA * v[t];
  hn[t] = x;
  if (bfout) bfout[t] = f2bf(x);
}

// ---------------- LayerNorm (one block per row) ----------------
__global__ __launch_bounds__(256) void k_layernorm(
    const float* __restrict__ y, const float* __restrict__ g,
    const float* __restrict__ b, float* __restrict__ out) {
  __shared__ float s1[256], s2[256];
  int row = blockIdx.x, tid = threadIdx.x;
  const float* yr = y + (size_t)row * Dd;
  float a0 = yr[tid], a1 = yr[tid + 256], a2 = yr[tid + 512];
  s1[tid] = a0 + a1 + a2;
  s2[tid] = a0 * a0 + a1 * a1 + a2 * a2;
  for (int st = 128; st > 0; st >>= 1) {
    __syncthreads();
    if (tid < st) { s1[tid] += s1[tid + st]; s2[tid] += s2[tid + st]; }
  }
  __syncthreads();
  float mu = s1[0] * (1.0f / Dd);
  float var = s2[0] * (1.0f / Dd) - mu * mu;
  float inv = rsqrtf(var + LN_EPS);
  float* orow = out + (size_t)row * Dd;
  orow[tid]       = (a0 - mu) * inv * g[tid]       + b[tid];
  orow[tid + 256] = (a1 - mu) * inv * g[tid + 256] + b[tid + 256];
  orow[tid + 512] = (a2 - mu) * inv * g[tid + 512] + b[tid + 512];
}

// ---------------- host orchestration ----------------
static inline int cdiv(int a, int b) { return (a + b - 1) / b; }

extern "C" void kernel_launch(void* const* d_in, const int* in_sizes, int n_in,
                              void* d_out, int out_size, void* d_ws, size_t ws_size,
                              hipStream_t stream) {
  const float* x    = (const float*)d_in[0];
  const int*   am   = (const int*)d_in[1];
  const int*   src  = (const int*)d_in[2];
  const int*   dst  = (const int*)d_in[3];
  const float* Wq   = (const float*)d_in[4];
  const float* bq   = (const float*)d_in[5];
  const float* Wk   = (const float*)d_in[6];
  const float* bk   = (const float*)d_in[7];
  const float* Wv   = (const float*)d_in[8];
  const float* bv   = (const float*)d_in[9];
  const float* Wo   = (const float*)d_in[10];
  const float* bo   = (const float*)d_in[11];
  const float* lng  = (const float*)d_in[12];
  const float* lnb  = (const float*)d_in[13];
  float* out = (float*)d_out;

  // workspace layout
  char* ws = (char*)d_ws;
  size_t off = 0;
  auto take = [&](size_t bytes) { char* p = ws + off; off += (bytes + 255) & ~(size_t)255; return p; };
  __bf16* xb   = (__bf16*)take((size_t)NN * Dd * 2);
  __bf16* WqT  = (__bf16*)take((size_t)Dd * Dd * 2);
  __bf16* WkT  = (__bf16*)take((size_t)Dd * Dd * 2);
  __bf16* WvT  = (__bf16*)take((size_t)Dd * Dd * 2);
  __bf16* WoT  = (__bf16*)take((size_t)Dd * Dd * 2);
  float*  qb   = (float*)take((size_t)NN * Dd * 4);   // reused as y-buffer later
  float*  kb   = (float*)take((size_t)NN * Dd * 4);
  float*  vb   = (float*)take((size_t)NN * Dd * 4);
  float*  attn = (float*)take((size_t)EE * Hh * 4);   // scores -> exp -> attn in place
  unsigned* smax = (unsigned*)take((size_t)NN * Hh * 4);
  float*  denom = (float*)take((size_t)NN * Hh * 4);
  float*  hA   = (float*)take((size_t)NN * Dd * 4);
  float*  hB   = (float*)take((size_t)NN * Dd * 4);
  __bf16* hb16 = (__bf16*)take((size_t)NN * Dd * 2);
  (void)ws_size; (void)n_in; (void)in_sizes; (void)out_size;

  const int ND = NN * Dd;          // 6,291,456
  const int EH = EE * Hh;          // 6,291,456
  const int DD = Dd * Dd;          // 589,824
  const int NH = NN * Hh;          // 98,304

  // 1) convert inputs to bf16 (x) and transposed-bf16 weights
  k_f32_to_bf16<<<cdiv(ND, 256), 256, 0, stream>>>(x, xb, ND);
  k_transpose_bf16<<<cdiv(DD, 256), 256, 0, stream>>>(Wq, WqT, Dd);
  k_transpose_bf16<<<cdiv(DD, 256), 256, 0, stream>>>(Wk, WkT, Dd);
  k_transpose_bf16<<<cdiv(DD, 256), 256, 0, stream>>>(Wv, WvT, Dd);
  k_transpose_bf16<<<cdiv(DD, 256), 256, 0, stream>>>(Wo, WoT, Dd);

  // 2) Q/K/V projections via WMMA (Q includes 1/sqrt(HD) scale)
  dim3 gg(Dd / 128, NN / 64);      // (6, 128)
  k_gemm_bf16<<<gg, 256, 0, stream>>>(xb, WqT, bq, nullptr, qb, NN, Dd, Dd, QSCALE);
  k_gemm_bf16<<<gg, 256, 0, stream>>>(xb, WkT, bk, nullptr, kb, NN, Dd, Dd, 1.0f);
  k_gemm_bf16<<<gg, 256, 0, stream>>>(xb, WvT, bv, nullptr, vb, NN, Dd, Dd, 1.0f);

  // 3) edge softmax over incoming edges of each dst node
  k_edge_scores<<<cdiv(EH, 256), 256, 0, stream>>>(qb, kb, src, dst, am, attn);
  k_fill_u32<<<cdiv(NH, 256), 256, 0, stream>>>(smax, 0x007FFFFFu /* encF(-inf) */, NH);
  k_edge_max<<<cdiv(EH, 256), 256, 0, stream>>>(attn, dst, smax);
  k_fill_f32<<<cdiv(NH, 256), 256, 0, stream>>>(denom, 0.0f, NH);
  k_edge_exp<<<cdiv(EH, 256), 256, 0, stream>>>(attn, dst, smax, denom);
  k_edge_norm<<<cdiv(EH, 256), 256, 0, stream>>>(attn, dst, denom);

  // 4) 5-step sparse diffusion: h0 = v; h <- (1-a)*segsum(attn*h[src]) + a*v
  const int SC = EE * 192;         // 100,663,296 scatter threads
  float* hcur = vb;
  float* bufs[2] = {hA, hB};
  for (int s = 0; s < N_STEPS; ++s) {
    float* hn = bufs[s & 1];
    k_fill_f32<<<cdiv(ND, 256), 256, 0, stream>>>(hn, 0.0f, ND);
    k_scatter<<<cdiv(SC, 256), 256, 0, stream>>>(attn, hcur, src, dst, hn);
    k_combine<<<cdiv(ND, 256), 256, 0, stream>>>(hn, vb, (s == N_STEPS - 1) ? hb16 : nullptr, ND);
    hcur = hn;
  }

  // 5) output projection + residual (WMMA), into y-buffer (aliases qb, now dead)
  float* ybuf = qb;
  k_gemm_bf16<<<gg, 256, 0, stream>>>(hb16, WoT, bo, x, ybuf, NN, Dd, Dd, 1.0f);

  // 6) LayerNorm -> d_out
  k_layernorm<<<NN, 256, 0, stream>>>(ybuf, lng, lnb, out);
}
